// DocAttention_34540126994938
// MI455X (gfx1250) — matrix operand order repaired and checked by reading, not compiled, not run on hardware
//
#include <hip/hip_runtime.h>
#include <hip/hip_bf16.h>

typedef __attribute__((ext_vector_type(16))) __bf16 bf16x16;
typedef __attribute__((ext_vector_type(8)))  float  v8f;

static constexpr int NB  = 32;    // batches
static constexpr int NA  = 32;    // docs per batch
static constexpr int SSQ = 128;   // seq len (only s=0 used)
static constexpr int DD  = 1024;  // hidden (K)
static constexpr int NHD = 1024;  // N*H flattened output cols
static constexpr int KST = DD / 32;    // 32 k-steps of K=32
static constexpr int CTI = NHD / 16;   // 64 column tiles of 16
static constexpr size_t WFRAG_ELEMS = (size_t)CTI * KST * 32 * 16; // 1M bf16 = 2 MiB/matrix

__device__ __forceinline__ __bf16 f2bf(float f) {
  union { float f; unsigned u; } v; v.f = f;
  unsigned r = v.u + 0x7FFFu + ((v.u >> 16) & 1u);   // round-to-nearest-even
  unsigned short s = (unsigned short)(r >> 16);
  __bf16 b; __builtin_memcpy(&b, &s, 2); return b;
}

// One-time: f32 weights (K x NHD, row-major) -> bf16 in B-fragment order
// dst[((ct*KST + ks)*32 + half*16 + colSub)*16 + e], K = ks*32 + half*16 + e
__global__ __launch_bounds__(256, 1) void convert_weights(
    const float* __restrict__ w, __bf16* __restrict__ dst) {
  int idx = blockIdx.x * 256 + threadIdx.x;        // 1M threads total
  int k = idx >> 10, col = idx & (NHD - 1);
  int ct = col >> 4, cs = col & 15;
  int ks = k >> 5,  kk = k & 31;
  int half = kk >> 4, e = kk & 15;
  dst[(((size_t)(ct * KST + ks) * 32) + half * 16 + cs) * 16 + e] = f2bf(w[idx]);
}

template <bool PRE>
__global__ __launch_bounds__(256, 1) void doc_att_fused(
    const float* __restrict__ enc, const int* __restrict__ dmask,
    const float* __restrict__ wqf, const float* __restrict__ bq,
    const float* __restrict__ wkf, const float* __restrict__ bk,
    const __bf16* __restrict__ wsq, const __bf16* __restrict__ wsk,
    float* __restrict__ out)
{
  // cls, bf16, pre-swizzled into WMMA A-fragment order: [mtile][kstep][lane][e]
  __shared__ __align__(32) __bf16 clsSw[2 * KST * 32 * 16];   // 64 KiB
  __shared__ float partials[8][NA];
  __shared__ float msh[NA];

  const int b    = blockIdx.x;
  const int tid  = threadIdx.x;
  const int lane = tid & 31;
  const int w    = tid >> 5;
  const bool hi  = lane >= 16;
  const int colSub = lane & 15;

  // ---- stage 1: load cls slice, convert to bf16, swizzle to A-frag layout ----
  for (int idx = tid; idx < NA * DD; idx += 256) {
    int a = idx >> 10, d = idx & (DD - 1);
    float f = enc[(size_t)(b * NA + a) * SSQ * DD + d];
    int mt = a >> 4, l0 = a & 15;
    int ks = d >> 5, kk = d & 31;
    int half = kk >> 4;                            // A-frag K contiguous per lane half
    int e    = kk & 15;
    clsSw[(((mt * KST + ks) * 32) + half * 16 + l0) * 16 + e] = f2bf(f);
  }
  if (tid < NA) msh[tid] = (float)dmask[b * NA + tid];
  __syncthreads();

  float plog[16];
  #pragma unroll
  for (int i = 0; i < 16; ++i) plog[i] = 0.f;

  // ---- stage 2: dual GEMM (q,k), fused logit accumulation ----
  for (int t = 0; t < 8; ++t) {
    const int ct  = w + t * 8;          // column tile 0..63, strided over waves
    const int col = ct * 16 + colSub;
    v8f cq0 = {}, cq1 = {}, ck0 = {}, ck1 = {};

    for (int ks = 0; ks < KST; ++ks) {
      bf16x16 a0 = *(const bf16x16*)&clsSw[((ks * 32) + lane) * 16];
      bf16x16 a1 = *(const bf16x16*)&clsSw[(((KST + ks) * 32) + lane) * 16];
      bf16x16 bfq, bfk;
      if constexpr (PRE) {
        const size_t fo = ((size_t)(ct * KST + ks) * 32 + lane) * 16;
        bfq = *(const bf16x16*)&wsq[fo];            // 32B contiguous per lane
        bfk = *(const bf16x16*)&wsk[fo];
        if (ks + 1 < KST) {
          __builtin_prefetch(&wsq[fo + 512], 0, 0); // next k-step fragment
          __builtin_prefetch(&wsk[fo + 512], 0, 0);
        }
      } else {
        const int kbase = ks * 32 + (hi ? 16 : 0);  // B-frag: lane half selects K group
        #pragma unroll
        for (int e = 0; e < 16; ++e) {
          size_t off = (size_t)(kbase + e) * NHD + col;
          bfq[e] = f2bf(wqf[off]);
          bfk[e] = f2bf(wkf[off]);
        }
      }
      cq0 = __builtin_amdgcn_wmma_f32_16x16x32_bf16(false, a0, false, bfq, (short)0, cq0, false, false);
      cq1 = __builtin_amdgcn_wmma_f32_16x16x32_bf16(false, a1, false, bfq, (short)0, cq1, false, false);
      ck0 = __builtin_amdgcn_wmma_f32_16x16x32_bf16(false, a0, false, bfk, (short)0, ck0, false, false);
      ck1 = __builtin_amdgcn_wmma_f32_16x16x32_bf16(false, a1, false, bfk, (short)0, ck1, false, false);
    }

    // bias + doc mask; C/D layout: VGPR r -> row a = r + 8*hi (+16 for mtile1), col = lane&15
    const float bqv = bq[col], bkv = bk[col];
    float qv0[8], kv0[8], qv1[8], kv1[8];
    float ksumL = 0.f;
    #pragma unroll
    for (int r = 0; r < 8; ++r) {
      float m0 = msh[r + (hi ? 8 : 0)];
      float m1 = msh[16 + r + (hi ? 8 : 0)];
      qv0[r] = (cq0[r] + bqv) * m0;  kv0[r] = (ck0[r] + bkv) * m0;
      qv1[r] = (cq1[r] + bqv) * m1;  kv1[r] = (ck1[r] + bkv) * m1;
      ksumL += kv0[r] + kv1[r];
    }
    // lane and lane^16 hold the same column, different row halves -> full column sum
    const float ksum = ksumL + __shfl_xor(ksumL, 16, 32);

    #pragma unroll
    for (int r = 0; r < 8; ++r) {
      // logits[x] += q[x,col]*(Ksum[col] - k[x,col])   (-k term removes diag y==x)
      plog[r]     += qv0[r] * (ksum - kv0[r]);   // x = r + 8*hi
      plog[8 + r] += qv1[r] * (ksum - kv1[r]);   // x = 16 + r + 8*hi
    }
  }

  // one deterministic butterfly over the 16 columns in each half (hoisted out of tile loop)
  #pragma unroll
  for (int r = 0; r < 16; ++r) {
    float v = plog[r];
    #pragma unroll
    for (int m = 1; m < 16; m <<= 1) v += __shfl_xor(v, m, 32);
    plog[r] = v;
  }

  if (colSub == 0) {                        // lanes 0 and 16 publish per-wave partials
    #pragma unroll
    for (int r = 0; r < 8; ++r) {
      partials[w][r + (hi ? 8 : 0)]      = plog[r];
      partials[w][16 + r + (hi ? 8 : 0)] = plog[8 + r];
    }
  }
  __syncthreads();

  // ---- stage 3: fixed-order cross-wave reduction + masked softmax (wave 0) ----
  if (tid < NA) {
    const int x = tid;
    float lg = 0.f;
    #pragma unroll
    for (int i = 0; i < 8; ++i) lg += partials[i][x];
    lg += (1.0f - msh[x]) * -100000.0f;
    float mx = lg;
    #pragma unroll
    for (int m = 16; m >= 1; m >>= 1) mx = fmaxf(mx, __shfl_xor(mx, m, 32));
    float ev = __expf(lg - mx);
    float sm = ev;
    #pragma unroll
    for (int m = 16; m >= 1; m >>= 1) sm += __shfl_xor(sm, m, 32);
    out[b * NA + x] = ev / sm;
  }
}

extern "C" void kernel_launch(void* const* d_in, const int* in_sizes, int n_in,
                              void* d_out, int out_size, void* d_ws, size_t ws_size,
                              hipStream_t stream) {
  (void)in_sizes; (void)n_in; (void)out_size;
  const float* enc = (const float*)d_in[0];
  const int*   dm  = (const int*)  d_in[1];
  const float* wq  = (const float*)d_in[2];
  const float* bq  = (const float*)d_in[3];
  const float* wk  = (const float*)d_in[4];
  const float* bk  = (const float*)d_in[5];
  float* out = (float*)d_out;

  const size_t need = 2 * WFRAG_ELEMS * sizeof(__bf16);   // 4 MiB
  if (d_ws != nullptr && ws_size >= need) {
    __bf16* wsq = (__bf16*)d_ws;
    __bf16* wsk = wsq + WFRAG_ELEMS;
    const int cvtBlocks = (int)((DD * (size_t)NHD) / 256);  // 4096
    convert_weights<<<cvtBlocks, 256, 0, stream>>>(wq, wsq);
    convert_weights<<<cvtBlocks, 256, 0, stream>>>(wk, wsk);
    doc_att_fused<true><<<NB, 256, 0, stream>>>(enc, dm, wq, bq, wk, bk, wsq, wsk, out);
  } else {
    doc_att_fused<false><<<NB, 256, 0, stream>>>(enc, dm, wq, bq, wk, bk, nullptr, nullptr, out);
  }
}
